// MultiHeadCrossAttention_40450001994146
// MI455X (gfx1250) — matrix operand order repaired and checked by reading, not compiled, not run on hardware
//
#include <hip/hip_runtime.h>
#include <hip/hip_bf16.h>
#include <math.h>

// ---------------------------------------------------------------------------
// MultiHeadCrossAttention for MI455X (gfx1250), bf16 WMMA with fp32 accumulate
// B=256, N=128, P=85(->96 used, 128 padded), D_LIG=256, D_PROT=1280,
// D_ATTN=256, H=4, dh=64
//
// Pipeline:
//   (a) one-time fp32->bf16 converts: lig (mask-zeroed), prot (pad-zeroed,
//       128-row pitch), W's pre-TRANSPOSED (Wt[n][k]) into workspace
//   (b) bf16 WMMA GEMMs with pure-copy LDS staging (no conversion in loop)
//   (c) fused attention (logits + mask/gate + softmax + attn@V)
//   (d) output projection with ligand-mask
// ---------------------------------------------------------------------------

typedef __bf16 v16bf __attribute__((ext_vector_type(16)));
typedef float  v8f   __attribute__((ext_vector_type(8)));

union Frag {
    v16bf v;
    unsigned u[8];
};

#if defined(__has_builtin)
#if __has_builtin(__builtin_amdgcn_cvt_pk_bf16_f32)
#define HAVE_CVT_PK_BF16 1
#endif
#endif

__device__ __forceinline__ unsigned pack2bf(float lo, float hi) {
#ifdef HAVE_CVT_PK_BF16
    typedef __bf16 v2bf __attribute__((ext_vector_type(2)));
    v2bf r = __builtin_amdgcn_cvt_pk_bf16_f32(lo, hi);
    return __builtin_bit_cast(unsigned, r);
#else
    unsigned ua = __builtin_bit_cast(unsigned, lo);
    unsigned ub = __builtin_bit_cast(unsigned, hi);
    ua = ua + 0x7FFFu + ((ua >> 16) & 1u);     // round-to-nearest-even
    ub = ub + 0x7FFFu + ((ub >> 16) & 1u);
    return (ua >> 16) | (ub & 0xFFFF0000u);
#endif
}
__device__ __forceinline__ unsigned short f2bfbits(float f) {
    unsigned u = __builtin_bit_cast(unsigned, f);
    unsigned r = u + 0x7FFFu + ((u >> 16) & 1u);
    return (unsigned short)(r >> 16);
}

// ---------------------------------------------------------------------------
// Convert kernel A: fp32 rows -> bf16 rows with row mask / pad-row zeroing.
// src: (batches, rows_valid, cols) fp32 ; dst: (batches, rows_pad, cols) bf16
// ---------------------------------------------------------------------------
__global__ __launch_bounds__(256)
void convert_rows_kernel(const float* __restrict__ src, unsigned short* __restrict__ dst,
                         const unsigned char* __restrict__ rowmask,
                         int rows_valid, int rows_pad, int cols, long total_f4)
{
    long g = (long)blockIdx.x * 256 + threadIdx.x;
    if (g >= total_f4) return;
    int cpr   = cols >> 2;                     // float4s per row
    int c4    = (int)(g % cpr) * 4;
    long rowg = g / cpr;                       // b * rows_pad + r
    int  b    = (int)(rowg / rows_pad);
    int  r    = (int)(rowg % rows_pad);
    bool valid = (r < rows_valid) &&
                 (rowmask == nullptr || rowmask[b * rows_valid + r] != 0);
    float4 v = make_float4(0.f, 0.f, 0.f, 0.f);
    if (valid) v = *(const float4*)(src + ((size_t)(b * rows_valid + r) * cols + c4));
    uint2 o;
    o.x = pack2bf(v.x, v.y);
    o.y = pack2bf(v.z, v.w);
    *(uint2*)(dst + (rowg * cols + c4)) = o;
}

// ---------------------------------------------------------------------------
// Convert kernel B: weight fp32 (Kd, Nout) -> bf16 TRANSPOSED Wt[n][k],
// packed as dwords of (k, k+1). Reads coalesced along n.
// ---------------------------------------------------------------------------
__global__ __launch_bounds__(256)
void convert_w_transpose_kernel(const float* __restrict__ W, unsigned* __restrict__ WtF32,
                                int Kd, int Nout, long total)
{
    long g = (long)blockIdx.x * 256 + threadIdx.x;
    if (g >= total) return;                    // total = Nout * (Kd/2)
    int n  = (int)(g % Nout);
    int k2 = (int)(g / Nout);
    float a = W[(size_t)(2 * k2)     * Nout + n];
    float b = W[(size_t)(2 * k2 + 1) * Nout + n];
    WtF32[(size_t)n * (Kd >> 1) + k2] = pack2bf(a, b);
}

// ---------------------------------------------------------------------------
// Kernel 1: Out[b, r, n] (bf16) = Abf[b, r, :] @ Wt[n, :]^T   (all bf16)
// 256 threads = 8 waves; block tile 64x64; wave tile 16x32 (2 WMMA / K-step).
// Double-buffered LDS; staging is pure uint2 copies.
// ---------------------------------------------------------------------------
__global__ __launch_bounds__(256)
void proj_gemm_kernel(const unsigned short* __restrict__ Abf,
                      const unsigned* __restrict__ WtF32,
                      unsigned short* __restrict__ Out,
                      int rows_pad, int Kd, int Nout, int tilesM)
{
    __shared__ unsigned short As[2][64 * 36];   // stride 36 ushorts = 18 dwords
    __shared__ unsigned short Wt[2][64 * 36];   // Wt[n][k] local tile

    const int tid  = threadIdx.x;
    const int b    = blockIdx.x / tilesM;
    const int mt   = blockIdx.x % tilesM;
    const int n0   = blockIdx.y * 64;
    const int w    = tid >> 5;
    const int lane = tid & 31;
    const int mloc = (w & 3) * 16;
    const int nloc = (w >> 2) * 32;
    const int fm   = lane & 15;
    const int fg   = lane >> 4;

    const unsigned* a32g = (const unsigned*)Abf;
    const int kdw = Kd >> 1;                    // dwords per bf16 row

    // loop-invariant source bases for this thread's two staging chunks
    const int er0 = (tid)       >> 3, ed0 = ((tid)       & 7) * 2;
    const int er1 = (tid + 256) >> 3, ed1 = ((tid + 256) & 7) * 2;
    const unsigned* srcA0 = a32g + (size_t)(b * rows_pad + mt * 64 + er0) * kdw + ed0;
    const unsigned* srcA1 = a32g + (size_t)(b * rows_pad + mt * 64 + er1) * kdw + ed1;
    const unsigned* srcW0 = WtF32 + (size_t)(n0 + er0) * kdw + ed0;
    const unsigned* srcW1 = WtF32 + (size_t)(n0 + er1) * kdw + ed1;

    auto stage = [&](int buf, int k0) {
        unsigned* as32 = (unsigned*)As[buf];
        unsigned* wt32 = (unsigned*)Wt[buf];
        int kd2 = k0 >> 1;
        *(uint2*)&as32[er0 * 18 + ed0] = *(const uint2*)(srcA0 + kd2);
        *(uint2*)&as32[er1 * 18 + ed1] = *(const uint2*)(srcA1 + kd2);
        *(uint2*)&wt32[er0 * 18 + ed0] = *(const uint2*)(srcW0 + kd2);
        *(uint2*)&wt32[er1 * 18 + ed1] = *(const uint2*)(srcW1 + kd2);
        if (k0 + 64 < Kd) {
            __builtin_prefetch(srcA0 + ((k0 + 64) >> 1), 0, 1);
            __builtin_prefetch(srcW0 + ((k0 + 64) >> 1), 0, 1);
        }
    };

    v8f c0 = {0.f,0.f,0.f,0.f,0.f,0.f,0.f,0.f};
    v8f c1 = {0.f,0.f,0.f,0.f,0.f,0.f,0.f,0.f};

    const int nk = Kd >> 5;
    stage(0, 0);
    for (int kt = 0; kt < nk; ++kt) {
        __syncthreads();
        if (kt + 1 < nk) stage((kt + 1) & 1, (kt + 1) * 32);

        const unsigned* as32 = (const unsigned*)As[kt & 1];
        const unsigned* wt32 = (const unsigned*)Wt[kt & 1];

        Frag a;
        int ab = (mloc + fm) * 18 + fg * 8;
        #pragma unroll
        for (int v = 0; v < 8; ++v) { a.u[v] = as32[ab + v]; }

        Frag bf0, bf1;
        int bb0 = (nloc + fm) * 18 + fg * 8;
        int bb1 = (nloc + 16 + fm) * 18 + fg * 8;
        #pragma unroll
        for (int v = 0; v < 8; ++v) { bf0.u[v] = wt32[bb0 + v]; bf1.u[v] = wt32[bb1 + v]; }

        c0 = __builtin_amdgcn_wmma_f32_16x16x32_bf16(false, a.v, false, bf0.v,
                                                     (short)0, c0, false, false);
        c1 = __builtin_amdgcn_wmma_f32_16x16x32_bf16(false, a.v, false, bf1.v,
                                                     (short)0, c1, false, false);
    }

    #pragma unroll
    for (int j = 0; j < 8; ++j) {
        int rloc = mt * 64 + mloc + j + 8 * fg;
        size_t rowbase = (size_t)(b * rows_pad + rloc) * Nout;
        Out[rowbase + n0 + nloc + fm]      = f2bfbits(c0[j]);
        Out[rowbase + n0 + nloc + 16 + fm] = f2bfbits(c1[j]);
    }
}

// ---------------------------------------------------------------------------
// Kernel 2: per (b, h, half-of-N) attention.  K/V buffers have 128-row pitch.
// A-fragments hoisted: built once, reused across all output tiles.
// ---------------------------------------------------------------------------
__global__ __launch_bounds__(128)
void attn_kernel(const unsigned short* __restrict__ Qbf,
                 const unsigned short* __restrict__ Kbf,
                 const unsigned short* __restrict__ Vbf,
                 const float* __restrict__ conf,
                 const unsigned char* __restrict__ pmask,
                 const float* __restrict__ conf_w, const float* __restrict__ conf_b,
                 float* __restrict__ attn_out, unsigned short* __restrict__ Ctx)
{
    __shared__ unsigned short Qs[64 * 68];    // Q rows (this half), stride 68
    __shared__ unsigned short Ks[96 * 68];    // K rows (96 used)
    __shared__ unsigned short Vts[64 * 100];  // V transposed: Vts[d][p], stride 100
    __shared__ float Ls[64 * 96];             // logits -> attn (fp32)
    __shared__ float addlog[96];

    const int tid  = threadIdx.x;
    const int bidx = blockIdx.x;
    const int b    = bidx >> 3;
    const int h    = (bidx >> 1) & 3;
    const int half = bidx & 1;

    unsigned* Qs32 = (unsigned*)Qs;
    unsigned* Ks32 = (unsigned*)Ks;
    unsigned* Vt32 = (unsigned*)Vts;

    const unsigned* Qg = (const unsigned*)Qbf;
    for (int idx = tid; idx < 64 * 32; idx += 128) {
        int r = idx >> 5, dw = idx & 31;
        Qs32[r * 34 + dw] = Qg[(size_t)(b * 128 + half * 64 + r) * 128 + h * 32 + dw];
    }
    const unsigned* Kg = (const unsigned*)Kbf;
    for (int idx = tid; idx < 96 * 32; idx += 128) {
        int r = idx >> 5, dw = idx & 31;
        Ks32[r * 34 + dw] = Kg[(size_t)(b * 128 + r) * 128 + h * 32 + dw];
    }
    // V transpose staging: 1 dword load -> 2 b16 stores
    const unsigned* Vg = (const unsigned*)Vbf;
    for (int idx = tid; idx < 96 * 32; idx += 128) {
        int p = idx >> 5, d2 = idx & 31;
        unsigned dv = Vg[(size_t)(b * 128 + p) * 128 + h * 32 + d2];
        int d = d2 * 2;
        Vts[d * 100 + p]       = (unsigned short)(dv & 0xFFFFu);
        Vts[(d + 1) * 100 + p] = (unsigned short)(dv >> 16);
    }
    if (tid < 96) {
        float v = -INFINITY;
        if (tid < 85 && !pmask[b * 85 + tid]) {
            float t  = conf[b * 85 + tid] * conf_w[0] + conf_b[0];
            float lg = -log1pf(expf(-t));            // log(sigmoid(t))
            v = fmaxf(lg, -10.0f);
        }
        addlog[tid] = v;
    }
    __syncthreads();

    const int w    = tid >> 5;
    const int lane = tid & 31;
    const int m0   = w * 16;
    const int fm   = lane & 15;
    const int fg   = lane >> 4;

    // ---- logits = Q K^T / 8 + addlog ----
    Frag aq[2];                                 // A-frags built once (kc only)
    #pragma unroll
    for (int kc = 0; kc < 2; ++kc) {
        int ab = (m0 + fm) * 34 + kc * 16 + fg * 8;
        #pragma unroll
        for (int v = 0; v < 8; ++v) {
            aq[kc].u[v]     = Qs32[ab + v];
        }
    }
    for (int pt = 0; pt < 6; ++pt) {
        v8f c = {0.f,0.f,0.f,0.f,0.f,0.f,0.f,0.f};
        #pragma unroll
        for (int kc = 0; kc < 2; ++kc) {
            Frag bf;
            int bb = (pt * 16 + fm) * 34 + kc * 16 + fg * 8;
            #pragma unroll
            for (int v = 0; v < 8; ++v) bf.u[v] = Ks32[bb + v];
            c = __builtin_amdgcn_wmma_f32_16x16x32_bf16(false, aq[kc].v, false, bf.v,
                                                        (short)0, c, false, false);
        }
        float al = addlog[pt * 16 + fm];
        #pragma unroll
        for (int j = 0; j < 8; ++j)
            Ls[(m0 + j + 8 * fg) * 96 + pt * 16 + fm] = c[j] * 0.125f + al;
    }
    __syncthreads();

    // ---- softmax over P ----
    if (tid < 64) {
        float* row = &Ls[tid * 96];
        float mx = -INFINITY;
        for (int p = 0; p < 96; ++p) mx = fmaxf(mx, row[p]);
        if (mx == -INFINITY) {
            for (int p = 0; p < 96; ++p) row[p] = 0.f;
        } else {
            float s = 0.f;
            for (int p = 0; p < 96; ++p) { float e = __expf(row[p] - mx); row[p] = e; s += e; }
            float r = 1.f / s;
            for (int p = 0; p < 96; ++p) row[p] *= r;
        }
        float* ao = attn_out + ((size_t)((b * 4 + h) * 128 + half * 64 + tid)) * 85;
        for (int p = 0; p < 85; ++p) ao[p] = row[p];
    }
    __syncthreads();

    // ---- ctx = attn @ V ----
    Frag aa[3];                                 // A-frags built+packed once
    #pragma unroll
    for (int kc = 0; kc < 3; ++kc) {
        const float* arow = &Ls[(m0 + fm) * 96 + kc * 32 + fg * 16];
        #pragma unroll
        for (int v = 0; v < 8; ++v) {
            aa[kc].u[v]     = pack2bf(arow[2 * v],      arow[2 * v + 1]);
        }
    }
    for (int nt = 0; nt < 4; ++nt) {
        v8f c = {0.f,0.f,0.f,0.f,0.f,0.f,0.f,0.f};
        #pragma unroll
        for (int kc = 0; kc < 3; ++kc) {
            Frag bf;
            int bb = (nt * 16 + fm) * 50 + kc * 16 + fg * 8;
            #pragma unroll
            for (int v = 0; v < 8; ++v) bf.u[v] = Vt32[bb + v];
            c = __builtin_amdgcn_wmma_f32_16x16x32_bf16(false, aa[kc].v, false, bf.v,
                                                        (short)0, c, false, false);
        }
        #pragma unroll
        for (int j = 0; j < 8; ++j)
            Ctx[(size_t)(b * 128 + half * 64 + m0 + j + 8 * fg) * 256
                + h * 64 + nt * 16 + fm] = f2bfbits(c[j]);
    }
}

// ---------------------------------------------------------------------------
// Kernel 3: out (fp32) = Ctx(bf16, 32768 x 256) @ WoT(bf16), * ligand_mask
// ---------------------------------------------------------------------------
__global__ __launch_bounds__(256)
void outproj_kernel(const unsigned short* __restrict__ Ctx,
                    const unsigned* __restrict__ WtF32,
                    const unsigned char* __restrict__ lmask,
                    float* __restrict__ Out)
{
    __shared__ unsigned short As[2][64 * 36];
    __shared__ unsigned short Wt[2][64 * 36];

    const int tid  = threadIdx.x;
    const int m0g  = blockIdx.x * 64;
    const int n0   = blockIdx.y * 64;
    const int w    = tid >> 5;
    const int lane = tid & 31;
    const int mloc = (w & 3) * 16;
    const int nloc = (w >> 2) * 32;
    const int fm   = lane & 15;
    const int fg   = lane >> 4;

    const unsigned* a32g = (const unsigned*)Ctx;

    const int er0 = (tid)       >> 3, ed0 = ((tid)       & 7) * 2;
    const int er1 = (tid + 256) >> 3, ed1 = ((tid + 256) & 7) * 2;
    const unsigned* srcA0 = a32g + (size_t)(m0g + er0) * 128 + ed0;
    const unsigned* srcA1 = a32g + (size_t)(m0g + er1) * 128 + ed1;
    const unsigned* srcW0 = WtF32 + (size_t)(n0 + er0) * 128 + ed0;
    const unsigned* srcW1 = WtF32 + (size_t)(n0 + er1) * 128 + ed1;

    auto stage = [&](int buf, int k0) {
        unsigned* as32 = (unsigned*)As[buf];
        unsigned* wt32 = (unsigned*)Wt[buf];
        int kd2 = k0 >> 1;
        *(uint2*)&as32[er0 * 18 + ed0] = *(const uint2*)(srcA0 + kd2);
        *(uint2*)&as32[er1 * 18 + ed1] = *(const uint2*)(srcA1 + kd2);
        *(uint2*)&wt32[er0 * 18 + ed0] = *(const uint2*)(srcW0 + kd2);
        *(uint2*)&wt32[er1 * 18 + ed1] = *(const uint2*)(srcW1 + kd2);
    };

    v8f c0 = {0.f,0.f,0.f,0.f,0.f,0.f,0.f,0.f};
    v8f c1 = {0.f,0.f,0.f,0.f,0.f,0.f,0.f,0.f};

    stage(0, 0);
    for (int kt = 0; kt < 8; ++kt) {
        __syncthreads();
        if (kt + 1 < 8) stage((kt + 1) & 1, (kt + 1) * 32);

        const unsigned* as32 = (const unsigned*)As[kt & 1];
        const unsigned* wt32 = (const unsigned*)Wt[kt & 1];

        Frag a;
        int ab = (mloc + fm) * 18 + fg * 8;
        #pragma unroll
        for (int v = 0; v < 8; ++v) { a.u[v] = as32[ab + v]; }

        Frag bf0, bf1;
        int bb0 = (nloc + fm) * 18 + fg * 8;
        int bb1 = (nloc + 16 + fm) * 18 + fg * 8;
        #pragma unroll
        for (int v = 0; v < 8; ++v) { bf0.u[v] = wt32[bb0 + v]; bf1.u[v] = wt32[bb1 + v]; }

        c0 = __builtin_amdgcn_wmma_f32_16x16x32_bf16(false, a.v, false, bf0.v,
                                                     (short)0, c0, false, false);
        c1 = __builtin_amdgcn_wmma_f32_16x16x32_bf16(false, a.v, false, bf1.v,
                                                     (short)0, c1, false, false);
    }

    #pragma unroll
    for (int j = 0; j < 8; ++j) {
        int grow = m0g + mloc + j + 8 * fg;
        float m  = lmask[grow] ? 1.f : 0.f;
        size_t rb = (size_t)grow * 256;
        Out[rb + n0 + nloc + fm]      = c0[j] * m;
        Out[rb + n0 + nloc + 16 + fm] = c1[j] * m;
    }
}

// ---------------------------------------------------------------------------
extern "C" void kernel_launch(void* const* d_in, const int* in_sizes, int n_in,
                              void* d_out, int out_size, void* d_ws, size_t ws_size,
                              hipStream_t stream)
{
    (void)in_sizes; (void)n_in; (void)out_size; (void)ws_size;

    const float*         lig   = (const float*)d_in[0];          // (256,128,256)
    const float*         prot  = (const float*)d_in[1];          // (256,85,1280)
    const float*         conf  = (const float*)d_in[2];          // (256,85)
    const unsigned char* pmask = (const unsigned char*)d_in[3];  // (256,85) bool
    const unsigned char* lmask = (const unsigned char*)d_in[4];  // (256,128) bool
    const float*         Wq    = (const float*)d_in[5];          // (256,256)
    const float*         Wk    = (const float*)d_in[6];          // (1280,256)
    const float*         Wv    = (const float*)d_in[7];          // (1280,256)
    const float*         Wo    = (const float*)d_in[8];          // (256,256)
    const float*         cw    = (const float*)d_in[9];
    const float*         cb    = (const float*)d_in[10];

    // ---- workspace layout (bf16, all offsets 8B-aligned), ~170 MB ----
    unsigned short* lig_bf  = (unsigned short*)d_ws;                 // 256*128*256
    unsigned short* prot_bf = lig_bf  + (size_t)256 * 128 * 256;     // 256*128*1280
    unsigned short* WqT     = prot_bf + (size_t)256 * 128 * 1280;    // 256*256
    unsigned short* WkT     = WqT     + (size_t)256 * 256;           // 256*1280
    unsigned short* WvT     = WkT     + (size_t)256 * 1280;          // 256*1280
    unsigned short* WoT     = WvT     + (size_t)256 * 1280;          // 256*256
    unsigned short* Qbf     = WoT     + (size_t)256 * 256;           // 256*128*256
    unsigned short* Kbf     = Qbf     + (size_t)256 * 128 * 256;
    unsigned short* Vbf     = Kbf     + (size_t)256 * 128 * 256;
    unsigned short* Ctx     = Vbf     + (size_t)256 * 128 * 256;

    float* out  = (float*)d_out;                                 // (256,128,256)
    float* attn = out + (size_t)256 * 128 * 256;                 // (256,4,128,85)

    // ---- one-time converts ----
    {
        long t1 = (long)256 * 128 * 64;        // lig: rows_pad=128, cols=256
        convert_rows_kernel<<<(unsigned)((t1 + 255) / 256), 256, 0, stream>>>(
            lig, lig_bf, lmask, 128, 128, 256, t1);
        long t2 = (long)256 * 128 * 320;       // prot: pad 85->128, cols=1280
        convert_rows_kernel<<<(unsigned)((t2 + 255) / 256), 256, 0, stream>>>(
            prot, prot_bf, nullptr, 85, 128, 1280, t2);
        long tq = (long)256 * 128;             // Wq: Nout*(Kd/2) = 256*128
        convert_w_transpose_kernel<<<(unsigned)((tq + 255) / 256), 256, 0, stream>>>(
            Wq, (unsigned*)WqT, 256, 256, tq);
        long tk = (long)256 * 640;             // Wk/Wv: 256*640
        convert_w_transpose_kernel<<<(unsigned)((tk + 255) / 256), 256, 0, stream>>>(
            Wk, (unsigned*)WkT, 1280, 256, tk);
        convert_w_transpose_kernel<<<(unsigned)((tk + 255) / 256), 256, 0, stream>>>(
            Wv, (unsigned*)WvT, 1280, 256, tk);
        convert_w_transpose_kernel<<<(unsigned)((tq + 255) / 256), 256, 0, stream>>>(
            Wo, (unsigned*)WoT, 256, 256, tq);
    }

    // ---- projections (bf16 GEMMs) ----
    proj_gemm_kernel<<<dim3(256 * 2, 4), 256, 0, stream>>>(
        lig_bf, (const unsigned*)WqT, Qbf, 128, 256, 256, 2);
    proj_gemm_kernel<<<dim3(256 * 2, 4), 256, 0, stream>>>(
        prot_bf, (const unsigned*)WkT, Kbf, 128, 1280, 256, 2);
    proj_gemm_kernel<<<dim3(256 * 2, 4), 256, 0, stream>>>(
        prot_bf, (const unsigned*)WvT, Vbf, 128, 1280, 256, 2);

    // ---- attention: one block per (b, h, N-half) ----
    attn_kernel<<<256 * 4 * 2, 128, 0, stream>>>(Qbf, Kbf, Vbf, conf, pmask,
                                                 cw, cb, attn, Ctx);

    // ---- output projection ----
    outproj_kernel<<<dim3(512, 4), 256, 0, stream>>>(
        Ctx, (const unsigned*)WoT, lmask, out);
}
